// SuperRes_69406671503911
// MI455X (gfx1250) — matrix operand order repaired and checked by reading, not compiled, not run on hardware
//
#include <hip/hip_runtime.h>

// ---------------------------------------------------------------------------
// DGCNN/EdgeConv block for MI455X (gfx1250, wave32, WMMA f16->f32).
// B=2, n=8192, ci=32, co=64, k=16.
// ---------------------------------------------------------------------------

typedef __attribute__((ext_vector_type(16))) _Float16 v16h;
typedef __attribute__((ext_vector_type(8)))  float    v8f;

#define B_    2
#define NPTS  8192
#define CI    32
#define CO    64
#define KNN   16
#define SPLIT 4              // candidate-axis split for the knn kernel

__device__ __forceinline__ v8f wmma_f16(v16h a, v16h b, v8f c) {
  // D = A(16x32 f16) * B(32x16 f16) + C(16x16 f32)
  return __builtin_amdgcn_wmma_f32_16x16x32_f16(false, a, false, b, (short)0, c,
                                                false, false);
}

__device__ __forceinline__ void lds_fence() {
  asm volatile("s_wait_dscnt 0" ::: "memory");
}

// Async global->LDS copy of 16 bytes per lane (ASYNCcnt-tracked).
__device__ __forceinline__ void async_copy_b128(void* lds, const void* gaddr) {
  unsigned l = (unsigned)(uintptr_t)lds;   // LDS address: low 32 bits of the pointer
  asm volatile("global_load_async_to_lds_b128 %0, %1, off"
               :: "v"(l), "v"(gaddr) : "memory");
}
__device__ __forceinline__ void async_wait() {
  asm volatile("s_wait_asynccnt 0" ::: "memory");
}

// A-matrix tile (16 rows x 32 K, f16) from row-major buffer (stride strideK
// halves). lane l -> row = l&15; positions 0..7 hold K = k0 + (l>>4)*8 + p ;
// positions 8..15 hold K = k0 + 16 + (l>>4)*8 + (p-8).
__device__ __forceinline__ v16h load_a(const _Float16* base, int r0,
                                       int strideK, int k0) {
  const int lane = threadIdx.x & 31;
  const _Float16* r =
      base + (size_t)(r0 + (lane & 15)) * strideK + k0 + ((lane >> 4) << 3);
  v16h a;
#pragma unroll
  for (int i = 0; i < 8; ++i) { a[i] = r[i]; a[i + 8] = r[i + 16]; }
  return a;
}

// B-matrix tile (32 K x 16 cols, f16) where B[k][n] = W[n][k] and W is
// row-major with stride strideK. lane l -> col = n0 + (l&15);
// positions 0..7 hold K = k0 + (l>>4)*8 + p ; positions 8..15 hold
// K = k0 + 16 + (l>>4)*8 + (p-8).
__device__ __forceinline__ v16h load_b(const _Float16* w, int n0,
                                       int strideK, int k0) {
  const int lane = threadIdx.x & 31;
  const _Float16* r =
      w + (size_t)(n0 + (lane & 15)) * strideK + k0 + ((lane >> 4) << 3);
  v16h b;
#pragma unroll
  for (int i = 0; i < 8; ++i) { b[i] = r[i]; b[i + 8] = r[i + 16]; }
  return b;
}

// ---------------------------------------------------------------------------
// Kernel 0: convert points to f16, compute |x|^2 per point.
// ---------------------------------------------------------------------------
__global__ __launch_bounds__(256) void prep_points(const float* __restrict__ feat,
                                                   _Float16* __restrict__ xh,
                                                   float* __restrict__ sq) {
  int p = blockIdx.x * 256 + threadIdx.x;
  if (p >= B_ * NPTS) return;
  float s = 0.f;
#pragma unroll
  for (int c = 0; c < CI; ++c) {
    float v = feat[(size_t)p * CI + c];
    s += v * v;
    xh[(size_t)p * CI + c] = (_Float16)v;
  }
  sq[p] = s;
}

// ---------------------------------------------------------------------------
// Kernel 1: convert all weights to f16, fold BN into scale/shift.
// ---------------------------------------------------------------------------
__global__ __launch_bounds__(256) void prep_weights(
    const float* ec_w1, const float* ec_g1, const float* ec_b1,
    const float* ec_m1, const float* ec_v1, const float* ec_w2,
    const float* fc1_w, const float* fc1_g, const float* fc1_b,
    const float* fc1_m, const float* fc1_v,
    const float* fc2_w, const float* fc2_g, const float* fc2_b,
    const float* fc2_m, const float* fc2_v,
    const float* fc3_w, const float* fc3_g, const float* fc3_b,
    const float* fc3_m, const float* fc3_v,
    const float* fc4_w,
    _Float16* w1h, _Float16* w2h, _Float16* f1h, _Float16* f2h,
    _Float16* f3h, _Float16* f4h,
    float* sc_e, float* sh_e, float* sc1, float* sh1,
    float* sc2, float* sh2, float* sc3, float* sh3) {
  const int tid = threadIdx.x;
  for (int i = tid; i < 128 * 64;  i += 256) w1h[i] = (_Float16)ec_w1[i];
  for (int i = tid; i < 64 * 128;  i += 256) w2h[i] = (_Float16)ec_w2[i];
  for (int i = tid; i < 192 * 224; i += 256) f1h[i] = (_Float16)fc1_w[i];
  for (int i = tid; i < 128 * 192; i += 256) f2h[i] = (_Float16)fc2_w[i];
  for (int i = tid; i < 64 * 128;  i += 256) f3h[i] = (_Float16)fc3_w[i];
  for (int i = tid; i < 64 * 64;   i += 256) f4h[i] = (_Float16)fc4_w[i];
  for (int i = tid; i < 128; i += 256) {
    float s = ec_g1[i] * rsqrtf(ec_v1[i] + 1e-5f);
    sc_e[i] = s; sh_e[i] = ec_b1[i] - ec_m1[i] * s;
  }
  for (int i = tid; i < 192; i += 256) {
    float s = fc1_g[i] * rsqrtf(fc1_v[i] + 1e-5f);
    sc1[i] = s; sh1[i] = fc1_b[i] - fc1_m[i] * s;
  }
  for (int i = tid; i < 128; i += 256) {
    float s = fc2_g[i] * rsqrtf(fc2_v[i] + 1e-5f);
    sc2[i] = s; sh2[i] = fc2_b[i] - fc2_m[i] * s;
  }
  for (int i = tid; i < 64; i += 256) {
    float s = fc3_g[i] * rsqrtf(fc3_v[i] + 1e-5f);
    sc3[i] = s; sh3[i] = fc3_b[i] - fc3_m[i] * s;
  }
}

// ---------------------------------------------------------------------------
// Kernel 2: fused pairwise-score + partial top-16. Each wave owns 32 query
// rows and 1/SPLIT of the candidates (2 WMMAs per 16-candidate tile).
// score = 2*dot - |x_j|^2 (-|x_i|^2 is row-constant: order-invariant).
// Emits a sorted (desc) partial top-16 list per (row, split).
// ---------------------------------------------------------------------------
__global__ __launch_bounds__(128) void knn_kernel(const _Float16* __restrict__ xh,
                                                  const float* __restrict__ sq,
                                                  float* __restrict__ pkv,
                                                  int* __restrict__ pki) {
  __shared__ float sc[4][32 * 17];
  const int lane = threadIdx.x & 31;
  const int wave = threadIdx.x >> 5;
  const int g    = blockIdx.x * 4 + wave;        // (row-group, split) pairs
  const int grp  = g / SPLIT;                    // 512 groups of 32 rows
  const int sp   = g % SPLIT;
  const int b    = grp / (NPTS / 32);
  const int i0   = (grp % (NPTS / 32)) * 32;
  const _Float16* xb  = xh + (size_t)b * NPTS * CI;
  const float*    sqb = sq + (size_t)b * NPTS;

  v16h a0 = load_a(xb, i0,      CI, 0);
  v16h a1 = load_a(xb, i0 + 16, CI, 0);

  float kv[KNN]; int ki[KNN];
#pragma unroll
  for (int t = 0; t < KNN; ++t) { kv[t] = -3.0e38f; ki[t] = 0; }

  const int myrow = i0 + lane;   // local row this lane owns for selection
  float* s = sc[wave];
  const int col = lane & 15;
  const int hs  = lane >> 4;

  const int jt0 = sp * (NPTS / 16 / SPLIT);
  const int jt1 = jt0 + (NPTS / 16 / SPLIT);
  for (int jt = jt0; jt < jt1; ++jt) {
    v16h bm = load_b(xb, jt * 16, CI, 0);
    v8f z = {};
    v8f c0 = wmma_f16(a0, bm, z);
    v8f c1 = wmma_f16(a1, bm, z);
    float sqj = sqb[jt * 16 + col];
#pragma unroll
    for (int r = 0; r < 8; ++r) {
      int m = r + (hs << 3);
      s[m * 17 + col]        = 2.0f * c0[r] - sqj;
      s[(m + 16) * 17 + col] = 2.0f * c1[r] - sqj;
    }
    lds_fence();
#pragma unroll
    for (int t = 0; t < 16; ++t) {
      int cand = jt * 16 + t;
      float v = s[lane * 17 + t];
      if (cand != myrow && v > kv[KNN - 1]) {
        float cv = v; int cj = cand;
#pragma unroll
        for (int q = 0; q < KNN; ++q) {   // sorted (desc) insert, branchless
          bool gt = cv > kv[q];
          float tv = kv[q]; int ti = ki[q];
          kv[q] = gt ? cv : tv; ki[q] = gt ? cj : ti;
          cv = gt ? tv : cv;    cj = gt ? ti : cj;
        }
      }
    }
  }
  size_t base = (((size_t)b * NPTS + myrow) * SPLIT + sp) * KNN;
#pragma unroll
  for (int t = 0; t < KNN; ++t) { pkv[base + t] = kv[t]; pki[base + t] = ki[t]; }
}

// ---------------------------------------------------------------------------
// Kernel 2b: merge SPLIT sorted partial lists per row into final top-16.
// Tie-break: larger value first, equal values -> smaller index first.
// ---------------------------------------------------------------------------
__global__ __launch_bounds__(256) void knn_merge(const float* __restrict__ pkv,
                                                 const int* __restrict__ pki,
                                                 int* __restrict__ knn) {
  int row = blockIdx.x * 256 + threadIdx.x;
  if (row >= B_ * NPTS) return;
  const float* kvb = pkv + (size_t)row * SPLIT * KNN;
  const int*   kib = pki + (size_t)row * SPLIT * KNN;
  int h0 = 0, h1 = 0, h2 = 0, h3 = 0;
#pragma unroll 1
  for (int t = 0; t < KNN; ++t) {
    float v0 = kvb[0 * KNN + h0], v1 = kvb[1 * KNN + h1];
    float v2 = kvb[2 * KNN + h2], v3 = kvb[3 * KNN + h3];
    int   i0 = kib[0 * KNN + h0], i1 = kib[1 * KNN + h1];
    int   i2 = kib[2 * KNN + h2], i3 = kib[3 * KNN + h3];
    // pairwise tournament with (value desc, index asc) order
    bool a01 = (v0 > v1) || (v0 == v1 && i0 < i1);
    float va = a01 ? v0 : v1; int ia = a01 ? i0 : i1; int sa = a01 ? 0 : 1;
    bool a23 = (v2 > v3) || (v2 == v3 && i2 < i3);
    float vb = a23 ? v2 : v3; int ib = a23 ? i2 : i3; int sb = a23 ? 2 : 3;
    bool ab  = (va > vb) || (va == vb && ia < ib);
    int sel  = ab ? sa : sb;
    knn[(size_t)row * KNN + t] = ab ? ia : ib;
    h0 += (sel == 0); h1 += (sel == 1); h2 += (sel == 2); h3 += (sel == 3);
  }
}

// ---------------------------------------------------------------------------
// Kernel 3: EdgeConv. One wave per point: ef(16x64) -> 16x128 (BN+lrelu)
// -> 16x64 (+bias) -> max over k. Weights staged block-wide into LDS via
// async global->LDS copies; all GEMMs via WMMA with hoisted A fragments.
// ---------------------------------------------------------------------------
__global__ __launch_bounds__(128) void edgeconv_kernel(
    const float* __restrict__ feat, const int* __restrict__ knn,
    const _Float16* __restrict__ w1h, const float* __restrict__ sc1,
    const float* __restrict__ sh1, const _Float16* __restrict__ w2h,
    const float* __restrict__ ec_b2, float* __restrict__ f1) {
  __shared__ _Float16 efA[4][16 * 64];     //  8 KB
  __shared__ _Float16 h1A[4][16 * 128];    // 16 KB
  __shared__ _Float16 w1s[128 * 64];       // 16 KB
  __shared__ _Float16 w2s[64 * 128];       // 16 KB
  const int tid  = threadIdx.x;
  const int lane = tid & 31;
  const int wave = tid >> 5;
  const int p = blockIdx.x * 4 + wave;     // global point id
  const int b = p / NPTS;
  const int row = lane & 15;
  const int hs  = lane >> 4;

  // Stage both weight matrices into LDS (async, 16B per lane per copy).
#pragma unroll
  for (int i = 0; i < (128 * 64) / (128 * 8); ++i) {     // 8 iters
    int off = (i * 128 + tid) * 8;
    async_copy_b128(w1s + off, w1h + off);
  }
#pragma unroll
  for (int i = 0; i < (64 * 128) / (128 * 8); ++i) {     // 8 iters
    int off = (i * 128 + tid) * 8;
    async_copy_b128(w2s + off, w2h + off);
  }

  int nb = knn[(size_t)p * KNN + row];
  const float* ctr = feat + (size_t)p * CI;
  const float* nbr = feat + ((size_t)b * NPTS + nb) * CI;
  _Float16* ef = efA[wave];
#pragma unroll
  for (int c = 0; c < 32; ++c) {
    float v = hs ? ctr[c] : (nbr[c] - ctr[c]);
    ef[row * 64 + (hs << 5) + c] = (_Float16)v;
  }
  async_wait();
  __syncthreads();
  lds_fence();

  v16h ea0 = load_a(ef, 0, 64, 0);
  v16h ea1 = load_a(ef, 0, 64, 32);
  _Float16* h1 = h1A[wave];
#pragma unroll 1
  for (int t = 0; t < 8; ++t) {                 // 128 outputs: 8 N-tiles
    v8f c = {};
    c = wmma_f16(ea0, load_b(w1s, t * 16, 64, 0),  c);
    c = wmma_f16(ea1, load_b(w1s, t * 16, 64, 32), c);
    int col = t * 16 + (lane & 15);
    float s = sc1[col], sh = sh1[col];
#pragma unroll
    for (int r = 0; r < 8; ++r) {
      float y = c[r] * s + sh;
      y = (y >= 0.f) ? y : 0.2f * y;
      h1[(r + (hs << 3)) * 128 + col] = (_Float16)y;
    }
  }
  lds_fence();

  v16h ha[4];
#pragma unroll
  for (int ks = 0; ks < 4; ++ks) ha[ks] = load_a(h1, 0, 128, ks * 32);
#pragma unroll 1
  for (int t = 0; t < 4; ++t) {                 // 64 outputs: 4 N-tiles
    v8f c = {};
#pragma unroll
    for (int ks = 0; ks < 4; ++ks)
      c = wmma_f16(ha[ks], load_b(w2s, t * 16, 128, ks * 32), c);
    int col = t * 16 + (lane & 15);
    float bb = ec_b2[col];
    float mx = -3.0e38f;
#pragma unroll
    for (int r = 0; r < 8; ++r) mx = fmaxf(mx, c[r] + bb);
    float oth = __shfl_xor(mx, 16, 32);         // rows 0-7 vs 8-15
    mx = fmaxf(mx, oth);
    if (lane < 16) f1[(size_t)p * CO + col] = mx;
  }
}

// ---------------------------------------------------------------------------
// Kernel 4: global max / mean over n per (batch, channel).
// ---------------------------------------------------------------------------
__global__ __launch_bounds__(256) void reduce_kernel(const float* __restrict__ f1,
                                                     float* __restrict__ fmx,
                                                     float* __restrict__ fav) {
  const int b = blockIdx.x >> 6;
  const int o = blockIdx.x & 63;
  __shared__ float smx[256], ssm[256];
  float mx = -3.0e38f, sm = 0.f;
  for (int i = threadIdx.x; i < NPTS; i += 256) {
    float v = f1[((size_t)b * NPTS + i) * CO + o];
    mx = fmaxf(mx, v); sm += v;
  }
  smx[threadIdx.x] = mx; ssm[threadIdx.x] = sm;
  __syncthreads();
  for (int st = 128; st > 0; st >>= 1) {
    if (threadIdx.x < st) {
      smx[threadIdx.x] = fmaxf(smx[threadIdx.x], smx[threadIdx.x + st]);
      ssm[threadIdx.x] += ssm[threadIdx.x + st];
    }
    __syncthreads();
  }
  if (threadIdx.x == 0) {
    fmx[b * CO + o] = smx[0];
    fav[b * CO + o] = ssm[0] / (float)NPTS;
  }
}

// ---------------------------------------------------------------------------
// Kernel 5: pointwise MLP 224->192->128->64->64, one wave per 16 points,
// WMMA chains with LDS ping-pong staging and hoisted A fragments.
// ---------------------------------------------------------------------------
__global__ __launch_bounds__(128) void mlp_kernel(
    const float* __restrict__ feat, const float* __restrict__ f1,
    const float* __restrict__ fmx, const float* __restrict__ fav,
    const _Float16* __restrict__ fc1h, const float* __restrict__ s1,
    const float* __restrict__ h1s,
    const _Float16* __restrict__ fc2h, const float* __restrict__ s2,
    const float* __restrict__ h2s,
    const _Float16* __restrict__ fc3h, const float* __restrict__ s3,
    const float* __restrict__ h3s,
    const _Float16* __restrict__ fc4h, const float* __restrict__ fc4b,
    float* __restrict__ out) {
  __shared__ _Float16 bufA[4][16 * 224];
  __shared__ _Float16 bufB[4][16 * 192];
  const int lane = threadIdx.x & 31;
  const int wave = threadIdx.x >> 5;
  const int p0 = (blockIdx.x * 4 + wave) * 16;
  const int b  = p0 / NPTS;
  const int hs = lane >> 4;
  _Float16* A  = bufA[wave];
  _Float16* Bm = bufB[wave];

  for (int t = lane; t < 16 * 224; t += 32) {   // stage cat = [feat|f1|max|avg]
    int r = t / 224, c = t % 224;
    float v;
    if (c < 32)       v = feat[((size_t)p0 + r) * CI + c];
    else if (c < 96)  v = f1[((size_t)p0 + r) * CO + (c - 32)];
    else if (c < 160) v = fmx[b * CO + (c - 96)];
    else              v = fav[b * CO + (c - 160)];
    A[t] = (_Float16)v;
  }
  lds_fence();

  v16h ak[7];
#pragma unroll
  for (int ks = 0; ks < 7; ++ks) ak[ks] = load_a(A, 0, 224, ks * 32);
#pragma unroll 1
  for (int t = 0; t < 12; ++t) {                // fc1: 224 -> 192
    v8f c = {};
#pragma unroll
    for (int ks = 0; ks < 7; ++ks)
      c = wmma_f16(ak[ks], load_b(fc1h, t * 16, 224, ks * 32), c);
    int col = t * 16 + (lane & 15);
    float s = s1[col], sh = h1s[col];
#pragma unroll
    for (int r = 0; r < 8; ++r) {
      float y = c[r] * s + sh; y = (y >= 0.f) ? y : 0.2f * y;
      Bm[(r + (hs << 3)) * 192 + col] = (_Float16)y;
    }
  }
  lds_fence();

#pragma unroll
  for (int ks = 0; ks < 6; ++ks) ak[ks] = load_a(Bm, 0, 192, ks * 32);
#pragma unroll 1
  for (int t = 0; t < 8; ++t) {                 // fc2: 192 -> 128 (into A)
    v8f c = {};
#pragma unroll
    for (int ks = 0; ks < 6; ++ks)
      c = wmma_f16(ak[ks], load_b(fc2h, t * 16, 192, ks * 32), c);
    int col = t * 16 + (lane & 15);
    float s = s2[col], sh = h2s[col];
#pragma unroll
    for (int r = 0; r < 8; ++r) {
      float y = c[r] * s + sh; y = (y >= 0.f) ? y : 0.2f * y;
      A[(r + (hs << 3)) * 128 + col] = (_Float16)y;
    }
  }
  lds_fence();

#pragma unroll
  for (int ks = 0; ks < 4; ++ks) ak[ks] = load_a(A, 0, 128, ks * 32);
#pragma unroll 1
  for (int t = 0; t < 4; ++t) {                 // fc3: 128 -> 64 (into Bm)
    v8f c = {};
#pragma unroll
    for (int ks = 0; ks < 4; ++ks)
      c = wmma_f16(ak[ks], load_b(fc3h, t * 16, 128, ks * 32), c);
    int col = t * 16 + (lane & 15);
    float s = s3[col], sh = h3s[col];
#pragma unroll
    for (int r = 0; r < 8; ++r) {
      float y = c[r] * s + sh; y = (y >= 0.f) ? y : 0.2f * y;
      Bm[(r + (hs << 3)) * 64 + col] = (_Float16)y;
    }
  }
  lds_fence();

#pragma unroll
  for (int ks = 0; ks < 2; ++ks) ak[ks] = load_a(Bm, 0, 64, ks * 32);
#pragma unroll 1
  for (int t = 0; t < 4; ++t) {                 // fc4: 64 -> 64 (+bias) -> out
    v8f c = {};
    c = wmma_f16(ak[0], load_b(fc4h, t * 16, 64, 0),  c);
    c = wmma_f16(ak[1], load_b(fc4h, t * 16, 64, 32), c);
    int col = t * 16 + (lane & 15);
    float bb = fc4b[col];
#pragma unroll
    for (int r = 0; r < 8; ++r)
      out[((size_t)p0 + r + (hs << 3)) * CO + col] = c[r] + bb;
  }
}

// ---------------------------------------------------------------------------
// Host launcher
// ---------------------------------------------------------------------------
extern "C" void kernel_launch(void* const* d_in, const int* in_sizes, int n_in,
                              void* d_out, int out_size, void* d_ws,
                              size_t ws_size, hipStream_t stream) {
  (void)in_sizes; (void)n_in; (void)out_size; (void)ws_size;
  const float* feat  = (const float*)d_in[0];
  const float* ec_w1 = (const float*)d_in[1];
  const float* ec_g1 = (const float*)d_in[2];
  const float* ec_b1 = (const float*)d_in[3];
  const float* ec_m1 = (const float*)d_in[4];
  const float* ec_v1 = (const float*)d_in[5];
  const float* ec_w2 = (const float*)d_in[6];
  const float* ec_b2 = (const float*)d_in[7];
  const float* fc1_w = (const float*)d_in[8];
  const float* fc1_g = (const float*)d_in[9];
  const float* fc1_b = (const float*)d_in[10];
  const float* fc1_m = (const float*)d_in[11];
  const float* fc1_v = (const float*)d_in[12];
  const float* fc2_w = (const float*)d_in[13];
  const float* fc2_g = (const float*)d_in[14];
  const float* fc2_b = (const float*)d_in[15];
  const float* fc2_m = (const float*)d_in[16];
  const float* fc2_v = (const float*)d_in[17];
  const float* fc3_w = (const float*)d_in[18];
  const float* fc3_g = (const float*)d_in[19];
  const float* fc3_b = (const float*)d_in[20];
  const float* fc3_m = (const float*)d_in[21];
  const float* fc3_v = (const float*)d_in[22];
  const float* fc4_w = (const float*)d_in[23];
  const float* fc4_b = (const float*)d_in[24];

  char* ws = (char*)d_ws;
  size_t off = 0;
  auto take = [&](size_t bytes) -> void* {
    void* p = (void*)(ws + off);
    off += (bytes + 255) & ~(size_t)255;
    return p;
  };

  _Float16* xh  = (_Float16*)take((size_t)B_ * NPTS * CI * 2);
  float*    sq  = (float*)take((size_t)B_ * NPTS * 4);
  int*      knn = (int*)take((size_t)B_ * NPTS * KNN * 4);
  float*    pkv = (float*)take((size_t)B_ * NPTS * SPLIT * KNN * 4);
  int*      pki = (int*)take((size_t)B_ * NPTS * SPLIT * KNN * 4);
  float*    f1  = (float*)take((size_t)B_ * NPTS * CO * 4);
  float*    fmx = (float*)take(B_ * CO * 4);
  float*    fav = (float*)take(B_ * CO * 4);
  _Float16* w1h = (_Float16*)take(128 * 64 * 2);
  _Float16* w2h = (_Float16*)take(64 * 128 * 2);
  _Float16* f1h = (_Float16*)take(192 * 224 * 2);
  _Float16* f2h = (_Float16*)take(128 * 192 * 2);
  _Float16* f3h = (_Float16*)take(64 * 128 * 2);
  _Float16* f4h = (_Float16*)take(64 * 64 * 2);
  float* sc_e = (float*)take(128 * 4); float* sh_e = (float*)take(128 * 4);
  float* sc1  = (float*)take(192 * 4); float* sh1  = (float*)take(192 * 4);
  float* sc2  = (float*)take(128 * 4); float* sh2  = (float*)take(128 * 4);
  float* sc3  = (float*)take(64 * 4);  float* sh3  = (float*)take(64 * 4);

  prep_points<<<(B_ * NPTS + 255) / 256, 256, 0, stream>>>(feat, xh, sq);
  prep_weights<<<1, 256, 0, stream>>>(
      ec_w1, ec_g1, ec_b1, ec_m1, ec_v1, ec_w2,
      fc1_w, fc1_g, fc1_b, fc1_m, fc1_v,
      fc2_w, fc2_g, fc2_b, fc2_m, fc2_v,
      fc3_w, fc3_g, fc3_b, fc3_m, fc3_v,
      fc4_w,
      w1h, w2h, f1h, f2h, f3h, f4h,
      sc_e, sh_e, sc1, sh1, sc2, sh2, sc3, sh3);
  knn_kernel<<<(B_ * NPTS / 32) * SPLIT / 4, 128, 0, stream>>>(xh, sq, pkv, pki);
  knn_merge<<<(B_ * NPTS + 255) / 256, 256, 0, stream>>>(pkv, pki, knn);
  edgeconv_kernel<<<(B_ * NPTS) / 4, 128, 0, stream>>>(feat, knn, w1h, sc_e,
                                                       sh_e, w2h, ec_b2, f1);
  reduce_kernel<<<B_ * CO, 256, 0, stream>>>(f1, fmx, fav);
  mlp_kernel<<<(B_ * NPTS / 16) / 4, 128, 0, stream>>>(
      feat, f1, fmx, fav, f1h, sc1, sh1, f2h, sc2, sh2, f3h, sc3, sh3, f4h,
      fc4_b, (float*)d_out);
}
